// RGCNLayer_35639638622237
// MI455X (gfx1250) — matrix operand order repaired and checked by reading, not compiled, not run on hardware
//
#include <hip/hip_runtime.h>
#include <cmath>

// ---------------------------------------------------------------------------
// RGCN layer with edge attention for gfx1250 (MI455X), bf16 WMMA mapping.
// ---------------------------------------------------------------------------

typedef __attribute__((ext_vector_type(16))) __bf16 v16bf;
typedef __attribute__((ext_vector_type(2)))  __bf16 v2bf;
typedef __attribute__((ext_vector_type(8)))  float  v8f;

#define INP   32
#define OUT   32
#define ATTN  32
#define NRELS 16
#define NBASE 8

#define WFRAGS 16               // 8 bases x 2 n-tiles
#define AFRAGS 4                // 2 A_w blocks (src, dst) x 2 n-tiles
#define TFRAGS (WFRAGS + AFRAGS)

static __device__ inline __bf16 f2bf(float f) {
  unsigned u = __builtin_bit_cast(unsigned, f);
  unsigned r = (u + 0x7FFFu + ((u >> 16) & 1u)) >> 16;   // round-to-nearest-even
  unsigned short s = (unsigned short)r;
  return __builtin_bit_cast(__bf16, s);
}

// packed f32 -> bf16 pair: hardware converter (confirmed selected on gfx1250)
static __device__ inline v2bf cvt2(float a, float b) {
#if __has_builtin(__builtin_amdgcn_cvt_pk_bf16_f32)
  return __builtin_amdgcn_cvt_pk_bf16_f32(a, b);
#else
  v2bf r;
  r[0] = (__bf16)a;
  r[1] = (__bf16)b;
  return r;
#endif
}

// A-matrix (16x32 bf16) per-lane element mapping (contiguous, matches B):
// lane L holds row M = L&15; K slots: j -> kb+j, kb=(L<16)?0:16
static __device__ inline void load_arow_f32(const float* __restrict__ row,
                                            float ha[16], int lane) {
  const int kb = (lane < 16) ? 0 : 16;
#pragma unroll
  for (int j = 0; j < 8; ++j) ha[j] = row[kb + j];
#pragma unroll
  for (int j = 0; j < 8; ++j) ha[8 + j] = row[kb + 8 + j];
}

static __device__ inline v16bf cvt_frag(const float ha[16]) {
  v16bf a;
#pragma unroll
  for (int j = 0; j < 16; j += 2) {
    v2bf p = cvt2(ha[j], ha[j + 1]);
    a[j] = p[0];
    a[j + 1] = p[1];
  }
  return a;
}

// scale a resident bf16 fragment by a scalar, staying in bf16
// (targets V_PK_MUL_BF16: one packed op per element pair)
static __device__ inline v16bf scale_frag_bf(v16bf a, float c) {
  v2bf cp = cvt2(c, c);
  v16bf r;
#pragma unroll
  for (int j = 0; j < 16; j += 2) {
    v2bf x;
    x[0] = a[j];
    x[1] = a[j + 1];
    v2bf y = x * cp;
    r[j] = y[0];
    r[j + 1] = y[1];
  }
  return r;
}

static __device__ inline v8f wmma_bf16(v16bf a, v16bf b, v8f c) {
  // D = A(16x32 bf16) x B(32x16 bf16) + C(16x16 f32)
  return __builtin_amdgcn_wmma_f32_16x16x32_bf16(false, a, false, b,
                                                 (short)0, c, false, false);
}

// ---------------------------------------------------------------------------
// Kernel 1: EAb[r][n] = attn_emb[r] @ A_w[64:96] + A_b   (16x32, one block)
// ---------------------------------------------------------------------------
__global__ void eab_kernel(const float* __restrict__ attn_emb,
                           const float* __restrict__ a_w,
                           const float* __restrict__ a_b,
                           float* __restrict__ eab) {
  int i = threadIdx.x;                  // 512 threads = 16*32
  int r = i >> 5, c = i & 31;
  float acc = a_b[c];
#pragma unroll
  for (int k = 0; k < ATTN; ++k)
    acc += attn_emb[r * ATTN + k] * a_w[(2 * INP + k) * ATTN + c];
  eab[i] = acc;
}

// ---------------------------------------------------------------------------
// Kernel 2: in-degree via f32 atomics
// ---------------------------------------------------------------------------
__global__ void degree_kernel(const int* __restrict__ dst,
                              float* __restrict__ indeg, int E) {
  int i = blockIdx.x * 256 + threadIdx.x;
  if (i < E) unsafeAtomicAdd(&indeg[dst[i]], 1.0f);
}

// ---------------------------------------------------------------------------
// Kernel 3: sum of log(indeg+1) for the scale mean
// ---------------------------------------------------------------------------
__global__ void logsum_kernel(const float* __restrict__ indeg,
                              float* __restrict__ sum_out, int Nn) {
  __shared__ float red[256];
  int i = blockIdx.x * 256 + threadIdx.x;
  float v = 0.0f;
  if (i < Nn) v = logf(indeg[i] + 1.0f);
  red[threadIdx.x] = v;
  __syncthreads();
#pragma unroll
  for (int s2 = 128; s2 > 0; s2 >>= 1) {
    if (threadIdx.x < s2) red[threadIdx.x] += red[threadIdx.x + s2];
    __syncthreads();
  }
  if (threadIdx.x == 0) unsafeAtomicAdd(sum_out, red[0]);
}

// ---------------------------------------------------------------------------
// Kernel 4: edge kernel — one 16-edge tile per wave iteration.
//   attention: Z = Hsrc@A1 + Hdst@A2 + EAb[etype]   (4 WMMAs)
//   message:   M = sum_b c_b * (Hsrc @ W_b)         (16 WMMAs, c folded into A)
//   a = sigmoid(relu(Z) @ B_w + B_b); atomics into agg[dst]
// ---------------------------------------------------------------------------
__global__ __launch_bounds__(256) void edge_kernel(
    const float* __restrict__ h, const float* __restrict__ weight,
    const float* __restrict__ w_comp, const float* __restrict__ a_w,
    const float* __restrict__ b_w, const float* __restrict__ b_b,
    const int* __restrict__ src, const int* __restrict__ dst,
    const int* __restrict__ e_type, const float* __restrict__ eab,
    float* __restrict__ agg, int E)
{
  // B-fragments pre-swizzled in LDS: frag f, lane l, slots j=0..15 contiguous
  // B-matrix mapping: lane l holds column N = nt*16 + (l&15), K = ((l<16)?0:16)+j
  __shared__ __attribute__((aligned(32))) __bf16 sFrag[TFRAGS * 512];
  __shared__ float sEAb[NRELS * ATTN];

  for (int idx = threadIdx.x; idx < TFRAGS * 512; idx += 256) {
    int f = idx >> 9, rem = idx & 511, l = rem >> 4, j = rem & 15;
    int col = l & 15, k = ((l < 16) ? 0 : 16) + j;
    float v;
    if (f < WFRAGS) {
      int b = f >> 1, nt = f & 1;
      v = weight[(b * INP + k) * OUT + nt * 16 + col];
    } else {
      int g = f - WFRAGS, br = g >> 1, nt = g & 1;
      v = a_w[(br * INP + k) * ATTN + nt * 16 + col];
    }
    sFrag[idx] = f2bf(v);
  }
  for (int idx = threadIdx.x; idx < NRELS * ATTN; idx += 256)
    sEAb[idx] = eab[idx];
  __syncthreads();

  const v16bf* fr = (const v16bf*)sFrag;

  const int lane  = threadIdx.x & 31;
  const int wave  = threadIdx.x >> 5;
  const int gwave = blockIdx.x * 8 + wave;
  const int nwaves = gridDim.x * 8;
  const int ntiles = (E + 15) >> 4;

  const int n0 = lane & 15, n1 = 16 + (lane & 15);
  const int hi = (lane >= 16) ? 1 : 0;
  const float Bw0 = b_w[n0], Bw1 = b_w[n1];
  const float Bb  = b_b[0];

  for (int tile = gwave; tile < ntiles; tile += nwaves) {
    const int m  = lane & 15;
    int e  = tile * 16 + m;
    int ec = e < E ? e : E - 1;
    const int s  = src[ec];
    const int d_ = dst[ec];
    const int t  = e_type[ec];

    // issue both row gathers up front, then convert (loads overlap)
    float hs[16], hd[16];
    load_arow_f32(h + (size_t)s  * INP, hs, lane);
    load_arow_f32(h + (size_t)d_ * INP, hd, lane);
    v16bf aS = cvt_frag(hs);
    v16bf aD = cvt_frag(hd);

    // --- attention GEMM ---
    v8f z0 = {}, z1 = {};
    z0 = wmma_bf16(aS, fr[(WFRAGS + 0) * 32 + lane], z0);
    z1 = wmma_bf16(aS, fr[(WFRAGS + 1) * 32 + lane], z1);
    z0 = wmma_bf16(aD, fr[(WFRAGS + 2) * 32 + lane], z0);
    z1 = wmma_bf16(aD, fr[(WFRAGS + 3) * 32 + lane], z1);

    // --- message GEMM: basis coefficients folded into scaled A-fragments.
    // unroll 1 keeps only one basis' B-fragments in flight (VGPR pressure);
    // the coefficient load is software-pipelined one iteration ahead so the
    // global-load latency hides under the WMMAs instead of stalling them.
    v8f m0 = {}, m1 = {};
    float c_cur = w_comp[t * NBASE];
#pragma unroll 1
    for (int b = 0; b < NBASE; ++b) {
      const int bn = (b + 1 < NBASE) ? (b + 1) : (NBASE - 1);
      const float c_next = w_comp[t * NBASE + bn];
      v16bf av = scale_frag_bf(aS, c_cur);
      m0 = wmma_bf16(av, fr[(b * 2 + 0) * 32 + lane], m0);
      m1 = wmma_bf16(av, fr[(b * 2 + 1) * 32 + lane], m1);
      c_cur = c_next;
    }

    // --- second attention layer: dot over N (cross-lane within 16-lane halves)
    float p[8];
#pragma unroll
    for (int r = 0; r < 8; ++r) {
      const int me = r + hi * 8;                 // edge row this lane covers
      const int te = __shfl(t, me, 32);
      const float z0v = z0[r] + sEAb[te * ATTN + n0];
      const float z1v = z1[r] + sEAb[te * ATTN + n1];
      p[r] = fmaxf(z0v, 0.0f) * Bw0 + fmaxf(z1v, 0.0f) * Bw1;
    }
#pragma unroll
    for (int r = 0; r < 8; ++r) {
#pragma unroll
      for (int mk = 1; mk < 16; mk <<= 1)
        p[r] += __shfl_xor(p[r], mk, 32);
    }

    // --- gated scatter into agg[dst] ---
#pragma unroll
    for (int r = 0; r < 8; ++r) {
      const int me = r + hi * 8;
      const int ee = tile * 16 + me;
      if (ee < E) {
        const float a = 1.0f / (1.0f + __expf(-(p[r] + Bb)));
        const int dd = __shfl(d_, me, 32);
        float* ap = agg + (size_t)dd * OUT;
        unsafeAtomicAdd(ap + n0, a * m0[r]);
        unsafeAtomicAdd(ap + n1, a * m1[r]);
      }
    }
  }
}

// ---------------------------------------------------------------------------
// Kernel 5: node epilogue — self-loop WMMA + scale/denom/bias + relu
// ---------------------------------------------------------------------------
__global__ __launch_bounds__(256) void node_kernel(
    const float* __restrict__ h, const float* __restrict__ selfw,
    const float* __restrict__ bias, const float* __restrict__ indeg,
    const float* __restrict__ logsum, const float* __restrict__ agg,
    float* __restrict__ out, int Nn)
{
  const int lane  = threadIdx.x & 31;
  const int wave  = threadIdx.x >> 5;
  const int gwave = blockIdx.x * 8 + wave;
  const int nwaves = gridDim.x * 8;
  const int ntiles = (Nn + 15) >> 4;

  const int n0 = lane & 15, n1 = 16 + n0;
  const int hi = (lane >= 16) ? 1 : 0;
  const int kb = (lane < 16) ? 0 : 16;

  v16bf w0, w1;                       // resident self-loop B fragments
#pragma unroll
  for (int j = 0; j < 16; ++j) {
    w0[j] = f2bf(selfw[(kb + j) * OUT + n0]);
    w1[j] = f2bf(selfw[(kb + j) * OUT + n1]);
  }
  const float invmean = (float)Nn / logsum[0];
  const float b0 = bias[n0], b1 = bias[n1];

  for (int tile = gwave; tile < ntiles; tile += nwaves) {
    const int m = lane & 15;
    int node = tile * 16 + m;
    int nc = node < Nn ? node : Nn - 1;
    float ha[16];
    load_arow_f32(h + (size_t)nc * INP, ha, lane);
    v16bf af = cvt_frag(ha);
    v8f c0 = {}, c1 = {};
    c0 = wmma_bf16(af, w0, c0);
    c1 = wmma_bf16(af, w1, c1);
#pragma unroll
    for (int r = 0; r < 8; ++r) {
      const int nodeR = tile * 16 + r + hi * 8;
      if (nodeR < Nn) {
        const float dg = indeg[nodeR];
        const float sc = logf(dg + 1.0f) * invmean;
        const float dn = fmaxf(dg, 1.0f);
        const float g0 = agg[(size_t)nodeR * OUT + n0];
        const float g1 = agg[(size_t)nodeR * OUT + n1];
        out[(size_t)nodeR * OUT + n0] = fmaxf(c0[r] + sc * g0 / dn + b0, 0.0f);
        out[(size_t)nodeR * OUT + n1] = fmaxf(c1[r] + sc * g1 / dn + b1, 0.0f);
      }
    }
  }
}

// ---------------------------------------------------------------------------
extern "C" void kernel_launch(void* const* d_in, const int* in_sizes, int n_in,
                              void* d_out, int out_size, void* d_ws, size_t ws_size,
                              hipStream_t stream) {
  const float* h        = (const float*)d_in[0];
  const float* weight   = (const float*)d_in[1];
  const float* w_comp   = (const float*)d_in[2];
  const float* selfw    = (const float*)d_in[3];
  const float* bias     = (const float*)d_in[4];
  const float* attn_emb = (const float*)d_in[5];
  const float* a_w      = (const float*)d_in[6];
  const float* a_b      = (const float*)d_in[7];
  const float* b_w      = (const float*)d_in[8];
  const float* b_b      = (const float*)d_in[9];
  const int*   src      = (const int*)d_in[10];
  const int*   dst      = (const int*)d_in[11];
  const int*   e_type   = (const int*)d_in[12];

  const int Nn = in_sizes[0] / INP;
  const int E  = in_sizes[10];

  float* wsf    = (float*)d_ws;
  float* eab    = wsf;                         // NRELS*ATTN = 512 floats
  float* indeg  = wsf + 512;                   // Nn floats
  float* lsum   = wsf + 512 + Nn;              // 1 float
  float* agg    = wsf + 512 + Nn + 1;          // Nn*OUT floats

  size_t zbytes = ((size_t)512 + (size_t)Nn + 1 + (size_t)Nn * OUT) * sizeof(float);
  hipMemsetAsync(d_ws, 0, zbytes, stream);

  eab_kernel   <<<1, 512, 0, stream>>>(attn_emb, a_w, a_b, eab);
  degree_kernel<<<(E + 255) / 256, 256, 0, stream>>>(dst, indeg, E);
  logsum_kernel<<<(Nn + 255) / 256, 256, 0, stream>>>(indeg, lsum, Nn);
  edge_kernel  <<<1024, 256, 0, stream>>>(h, weight, w_comp, a_w, b_w, b_b,
                                          src, dst, e_type, eab, agg, E);
  node_kernel  <<<784, 256, 0, stream>>>(h, selfw, bias, indeg, lsum, agg,
                                         (float*)d_out, Nn);
}
